// MoELayer_77945066488043
// MI455X (gfx1250) — matrix operand order repaired and checked by reading, not compiled, not run on hardware
//
#include <hip/hip_runtime.h>
#include <math.h>

#define D_MODEL 1024
#define D_FF    2048
#define N_EXP   8
#define T_TOK   8192
#define ZLOSS_COEF 0.001f

typedef __attribute__((ext_vector_type(16))) __bf16        v16bf;
typedef __attribute__((ext_vector_type(8)))  float         v8f;
typedef __attribute__((ext_vector_type(8)))  unsigned int  v8u;

// LDS tile row stride in ushorts: 32 K-values + 8 pad.
// Row stride = 80 bytes -> multiple of 16B (keeps ds_load_b128 aligned),
// and shifts bank mapping by 20 banks per row.
#define TSTR 40

__device__ __forceinline__ v16bf frag_cat(uint4 c0, uint4 c1) {
    v8u r;
    r[0] = c0.x; r[1] = c0.y; r[2] = c0.z; r[3] = c0.w;
    r[4] = c1.x; r[5] = c1.y; r[6] = c1.z; r[7] = c1.w;
    return __builtin_bit_cast(v16bf, r);
}

// A-matrix (16x32 bf16) fragment per ISA 7.12.2:
// lane = half*16 + lr, VGPR v holds K = (v>>2)*16 + half*8 + (v&3)*2 for row M=lr.
__device__ __forceinline__ v16bf frag_A(const unsigned short* t, int row, int half) {
    const unsigned short* p = t + row * TSTR;
    uint4 c0 = *(const uint4*)(p + half * 8);
    uint4 c1 = *(const uint4*)(p + 16 + half * 8);
    return frag_cat(c0, c1);
}
// B-matrix (32x16 bf16) fragment: lane = half*16 + n, VGPR v holds K = (v>>2)*16 + half*8 + (v&3)*2.
__device__ __forceinline__ v16bf frag_B(const unsigned short* t, int row, int half) {
    const unsigned short* p = t + row * TSTR;
    uint4 c0 = *(const uint4*)(p + half * 8);
    uint4 c1 = *(const uint4*)(p + 16 + half * 8);
    return frag_cat(c0, c1);
}

__device__ __forceinline__ unsigned int pk_bf16(float a, float b) {
    unsigned short ua = __builtin_bit_cast(unsigned short, (__bf16)a);
    unsigned short ub = __builtin_bit_cast(unsigned short, (__bf16)b);
    return (unsigned int)ua | ((unsigned int)ub << 16);
}

__device__ __forceinline__ void gload16(float4 r[4], const float* g) {
    r[0] = *(const float4*)(g + 0);
    r[1] = *(const float4*)(g + 4);
    r[2] = *(const float4*)(g + 8);
    r[3] = *(const float4*)(g + 12);
}
__device__ __forceinline__ void cvt_store16(unsigned short* l, const float4 r[4]) {
    *(uint4*)(l)     = make_uint4(pk_bf16(r[0].x, r[0].y), pk_bf16(r[0].z, r[0].w),
                                  pk_bf16(r[1].x, r[1].y), pk_bf16(r[1].z, r[1].w));
    *(uint4*)(l + 8) = make_uint4(pk_bf16(r[2].x, r[2].y), pk_bf16(r[2].z, r[2].w),
                                  pk_bf16(r[3].x, r[3].y), pk_bf16(r[3].z, r[3].w));
}
__device__ __forceinline__ void gload8(float4 r[2], const float* g) {
    r[0] = *(const float4*)(g + 0);
    r[1] = *(const float4*)(g + 4);
}
__device__ __forceinline__ void cvt_store8(unsigned short* l, const float4 r[2]) {
    *(uint4*)(l) = make_uint4(pk_bf16(r[0].x, r[0].y), pk_bf16(r[0].z, r[0].w),
                              pk_bf16(r[1].x, r[1].y), pk_bf16(r[1].z, r[1].w));
}

// Async DMA: global (bf16, 16B) -> LDS, tracked by ASYNCcnt.
// LDS byte offset = low 32 bits of the flat address (ISA 10.2 LDS aperture).
__device__ __forceinline__ void async_b128(unsigned short* l, const unsigned short* g) {
    asm volatile("global_load_async_to_lds_b128 %0, %1, off"
                 :: "v"((unsigned)(size_t)l), "v"(g)
                 : "memory");
}
__device__ __forceinline__ void wait_async0() {
    asm volatile("s_wait_asynccnt 0x0" ::: "memory");
}

// ---------------------------------------------------------------------------
// Kernel 1: per-expert  h = silu(x @ W1^T) * (x @ W3^T), stored bf16.
// C-tile 128(M) x 64(N); 8 waves as 4x2 grid, each wave a 32x32 sub-tile of
// BOTH GEMMs (shared A tile). Double-buffered LDS; next K-tile is fetched to
// registers before the barrier and converted/stored after the WMMAs.
// ---------------------------------------------------------------------------
__global__ __launch_bounds__(256) void ffn_h_kernel(
    const float* __restrict__ x,
    const float* __restrict__ W1e,
    const float* __restrict__ W3e,
    unsigned short* __restrict__ hbuf)
{
    __shared__ unsigned short sA [2][128 * TSTR];
    __shared__ unsigned short sB1[2][ 64 * TSTR];
    __shared__ unsigned short sB3[2][ 64 * TSTR];

    const int tid  = threadIdx.x;
    const int lane = tid & 31;
    const int wave = tid >> 5;        // 0..7
    const int wm   = wave >> 1;       // 0..3  (M sub-tile)
    const int wn   = wave & 1;        // 0..1  (N sub-tile)
    const int half = lane >> 4;
    const int lr   = lane & 15;

    const int bm = blockIdx.x;        // 128-token block
    const int bn = blockIdx.y;        // 64-ff block

    // staging assignments
    const int arow = tid >> 1;              // 0..127
    const int acol = (tid & 1) << 4;        // 0 / 16
    const int brow = (tid & 127) >> 1;      // 0..63
    const float* Wp = (tid & 128) ? W3e : W1e;

    const v8f vzero = {0.f, 0.f, 0.f, 0.f, 0.f, 0.f, 0.f, 0.f};
    v8f acc1[2][2], acc3[2][2];
#pragma unroll
    for (int i = 0; i < 2; ++i)
#pragma unroll
        for (int j = 0; j < 2; ++j) { acc1[i][j] = vzero; acc3[i][j] = vzero; }

    const float* gA = x  + (size_t)(bm * 128 + arow) * D_MODEL + acol;
    const float* gB = Wp + (size_t)(bn *  64 + brow) * D_MODEL + acol;
    unsigned short* lA[2] = { &sA[0][arow * TSTR + acol], &sA[1][arow * TSTR + acol] };
    unsigned short* lB[2];
    if (tid & 128) { lB[0] = &sB3[0][brow * TSTR + acol]; lB[1] = &sB3[1][brow * TSTR + acol]; }
    else           { lB[0] = &sB1[0][brow * TSTR + acol]; lB[1] = &sB1[1][brow * TSTR + acol]; }

    const int rA0 = wm * 32 + lr, rA1 = rA0 + 16;
    const int rB0 = wn * 32 + lr, rB1 = rB0 + 16;

    // prologue: stage K-tile 0
    float4 ra[4], rb[4];
    gload16(ra, gA);
    gload16(rb, gB);
    cvt_store16(lA[0], ra);
    cvt_store16(lB[0], rb);

    const int KSTEPS = D_MODEL / 32;   // 32
    for (int kb = 0; kb < KSTEPS; ++kb) {
        const int cur = kb & 1, nxt = cur ^ 1;
        const bool more = (kb + 1) < KSTEPS;
        if (more) {
            gload16(ra, gA + (kb + 1) * 32);   // into registers: no LDS hazard
            gload16(rb, gB + (kb + 1) * 32);
            if (kb + 2 < KSTEPS) __builtin_prefetch(gB + (kb + 2) * 32, 0, 0);
        }
        __syncthreads();                        // buffer `cur` is published

        v16bf a0 = frag_A(sA[cur],  rA0, half);
        v16bf a1 = frag_A(sA[cur],  rA1, half);
        v16bf p0 = frag_B(sB1[cur], rB0, half);
        v16bf p1 = frag_B(sB1[cur], rB1, half);
        v16bf q0 = frag_B(sB3[cur], rB0, half);
        v16bf q1 = frag_B(sB3[cur], rB1, half);

        acc1[0][0] = __builtin_amdgcn_wmma_f32_16x16x32_bf16(false, a0, false, p0, (short)0, acc1[0][0], false, false);
        acc1[0][1] = __builtin_amdgcn_wmma_f32_16x16x32_bf16(false, a0, false, p1, (short)0, acc1[0][1], false, false);
        acc1[1][0] = __builtin_amdgcn_wmma_f32_16x16x32_bf16(false, a1, false, p0, (short)0, acc1[1][0], false, false);
        acc1[1][1] = __builtin_amdgcn_wmma_f32_16x16x32_bf16(false, a1, false, p1, (short)0, acc1[1][1], false, false);
        acc3[0][0] = __builtin_amdgcn_wmma_f32_16x16x32_bf16(false, a0, false, q0, (short)0, acc3[0][0], false, false);
        acc3[0][1] = __builtin_amdgcn_wmma_f32_16x16x32_bf16(false, a0, false, q1, (short)0, acc3[0][1], false, false);
        acc3[1][0] = __builtin_amdgcn_wmma_f32_16x16x32_bf16(false, a1, false, q0, (short)0, acc3[1][0], false, false);
        acc3[1][1] = __builtin_amdgcn_wmma_f32_16x16x32_bf16(false, a1, false, q1, (short)0, acc3[1][1], false, false);

        if (more) {                             // stage K-tile kb+1 into `nxt`
            cvt_store16(lA[nxt], ra);
            cvt_store16(lB[nxt], rb);
        }
    }

    // epilogue: silu(h1) * h3 -> bf16 hbuf  (C layout: m = v + half*8, n = lr)
#pragma unroll
    for (int i = 0; i < 2; ++i)
#pragma unroll
        for (int j = 0; j < 2; ++j)
#pragma unroll
            for (int v = 0; v < 8; ++v) {
                const int mg = bm * 128 + wm * 32 + i * 16 + half * 8 + v;
                const int ng = bn *  64 + wn * 32 + j * 16 + lr;
                const float a1v = acc1[i][j][v];
                const float a3v = acc3[i][j][v];
                const float sig = __builtin_amdgcn_rcpf(1.0f + __expf(-a1v));
                const float hv  = a1v * sig * a3v;
                hbuf[(size_t)mg * D_FF + ng] =
                    __builtin_bit_cast(unsigned short, (__bf16)hv);
            }
}

// ---------------------------------------------------------------------------
// Kernel 2: per-expert  y = h @ W2^T ; out (+)= gate[t,e] * y
// A tile (bf16 in global) is DMA'd with GLOBAL_LOAD_ASYNC_TO_LDS_B128
// (ASYNCcnt), issued right after the barrier so it overlaps the WMMAs of the
// current tile; W2 tile is register-staged + converted as in kernel 1.
// ---------------------------------------------------------------------------
__global__ __launch_bounds__(256) void ffn_y_kernel(
    const unsigned short* __restrict__ hbuf,
    const float* __restrict__ W2e,
    const float* __restrict__ gates,
    float* __restrict__ out,
    int expert, int first)
{
    __shared__ unsigned short sA[2][128 * TSTR];
    __shared__ unsigned short sB[2][ 64 * TSTR];

    const int tid  = threadIdx.x;
    const int lane = tid & 31;
    const int wave = tid >> 5;
    const int wm   = wave >> 1;
    const int wn   = wave & 1;
    const int half = lane >> 4;
    const int lr   = lane & 15;

    const int bm = blockIdx.x;        // 128-token block
    const int bn = blockIdx.y;        // 64-d_model block

    const int arow = tid >> 1;              // 0..127
    const int acol = (tid & 1) << 4;        // 0 / 16
    const int brow = tid >> 2;              // 0..63
    const int bcol = (tid & 3) << 3;        // 0,8,16,24

    const v8f vzero = {0.f, 0.f, 0.f, 0.f, 0.f, 0.f, 0.f, 0.f};
    v8f acc[2][2];
#pragma unroll
    for (int i = 0; i < 2; ++i)
#pragma unroll
        for (int j = 0; j < 2; ++j) acc[i][j] = vzero;

    const unsigned short* gA = hbuf + (size_t)(bm * 128 + arow) * D_FF + acol;
    const float*          gB = W2e  + (size_t)(bn *  64 + brow) * D_FF + bcol;
    unsigned short* lA[2] = { &sA[0][arow * TSTR + acol], &sA[1][arow * TSTR + acol] };
    unsigned short* lB[2] = { &sB[0][brow * TSTR + bcol], &sB[1][brow * TSTR + bcol] };

    const int rA0 = wm * 32 + lr, rA1 = rA0 + 16;
    const int rB0 = wn * 32 + lr, rB1 = rB0 + 16;

    // prologue: stage K-tile 0 (A via async DMA, B via registers)
    async_b128(lA[0],     gA);
    async_b128(lA[0] + 8, gA + 8);
    float4 rb[2];
    gload8(rb, gB);
    cvt_store8(lB[0], rb);

    const int KSTEPS = D_FF / 32;   // 64
    for (int kb = 0; kb < KSTEPS; ++kb) {
        const int cur = kb & 1, nxt = cur ^ 1;
        const bool more = (kb + 1) < KSTEPS;
        if (more) {
            gload8(rb, gB + (kb + 1) * 32);     // registers: no LDS hazard
            if (kb + 2 < KSTEPS) __builtin_prefetch(gB + (kb + 2) * 32, 0, 0);
        }
        wait_async0();                          // tile kb's DMA landed in LDS
        __syncthreads();                        // buffer `cur` is published
        if (more) {                             // DMA tile kb+1; overlaps WMMAs
            async_b128(lA[nxt],     gA + (kb + 1) * 32);
            async_b128(lA[nxt] + 8, gA + (kb + 1) * 32 + 8);
        }

        v16bf a0 = frag_A(sA[cur], rA0, half);
        v16bf a1 = frag_A(sA[cur], rA1, half);
        v16bf b0 = frag_B(sB[cur], rB0, half);
        v16bf b1 = frag_B(sB[cur], rB1, half);

        acc[0][0] = __builtin_amdgcn_wmma_f32_16x16x32_bf16(false, a0, false, b0, (short)0, acc[0][0], false, false);
        acc[0][1] = __builtin_amdgcn_wmma_f32_16x16x32_bf16(false, a0, false, b1, (short)0, acc[0][1], false, false);
        acc[1][0] = __builtin_amdgcn_wmma_f32_16x16x32_bf16(false, a1, false, b0, (short)0, acc[1][0], false, false);
        acc[1][1] = __builtin_amdgcn_wmma_f32_16x16x32_bf16(false, a1, false, b1, (short)0, acc[1][1], false, false);

        if (more) cvt_store8(lB[nxt], rb);
    }

#pragma unroll
    for (int i = 0; i < 2; ++i)
#pragma unroll
        for (int v = 0; v < 8; ++v) {
            const int mg  = bm * 128 + wm * 32 + i * 16 + half * 8 + v;
            const float g = gates[(size_t)mg * N_EXP + expert];
#pragma unroll
            for (int j = 0; j < 2; ++j) {
                const int ng = bn * 64 + wn * 32 + j * 16 + lr;
                const size_t o = (size_t)mg * D_MODEL + ng;
                const float prev = first ? 0.0f : out[o];
                out[o] = prev + g * acc[i][j][v];
            }
        }
}

// ---------------------------------------------------------------------------
// Router: logits, softmax, top-2 gates; deterministic per-block partials of
// importance[8], load[8], sum(z^2).
// ---------------------------------------------------------------------------
__device__ __forceinline__ float block_reduce(float v, float* sRed) {
    const int tid = threadIdx.x;
    sRed[tid] = v;
    __syncthreads();
    for (int s = 128; s > 0; s >>= 1) {
        if (tid < s) sRed[tid] += sRed[tid + s];
        __syncthreads();
    }
    const float r = sRed[0];
    __syncthreads();
    return r;
}

__global__ __launch_bounds__(256) void router_kernel(
    const float* __restrict__ x,
    const float* __restrict__ Wr,
    float* __restrict__ gates,
    float* __restrict__ partials)
{
    __shared__ float sW[N_EXP * D_MODEL];   // 32 KB
    __shared__ float sRed[256];

    const int tid = threadIdx.x;
    for (int i = tid; i < N_EXP * D_MODEL; i += 256) sW[i] = Wr[i];
    __syncthreads();

    const int t = blockIdx.x * 256 + tid;   // token (T = 8192 = 32 * 256)
    const float* xr = x + (size_t)t * D_MODEL;

    float logit[N_EXP];
#pragma unroll
    for (int e = 0; e < N_EXP; ++e) logit[e] = 0.0f;
    for (int d = 0; d < D_MODEL; ++d) {
        const float xv = xr[d];
#pragma unroll
        for (int e = 0; e < N_EXP; ++e) logit[e] += xv * sW[e * D_MODEL + d];
    }

    float mx = logit[0];
#pragma unroll
    for (int e = 1; e < N_EXP; ++e) mx = fmaxf(mx, logit[e]);
    float p[N_EXP], se = 0.0f;
#pragma unroll
    for (int e = 0; e < N_EXP; ++e) { p[e] = __expf(logit[e] - mx); se += p[e]; }
    const float inv = 1.0f / se;
#pragma unroll
    for (int e = 0; e < N_EXP; ++e) p[e] *= inv;

    // top-2 (ties -> lowest index, matching top_k)
    int i1 = 0;
#pragma unroll
    for (int e = 1; e < N_EXP; ++e) if (p[e] > p[i1]) i1 = e;
    int i2 = (i1 == 0) ? 1 : 0;
#pragma unroll
    for (int e = 0; e < N_EXP; ++e) if (e != i1 && p[e] > p[i2]) i2 = e;

    const float s2 = p[i1] + p[i2] + 1e-9f;
#pragma unroll
    for (int e = 0; e < N_EXP; ++e) {
        float g = 0.0f;
        if (e == i1) g = p[i1] / s2;
        else if (e == i2) g = p[i2] / s2;
        gates[(size_t)t * N_EXP + e] = g;
    }

    const float z = __logf(se) + mx;

    const int bofs = blockIdx.x * 32;
#pragma unroll
    for (int e = 0; e < N_EXP; ++e) {
        const float s = block_reduce(p[e], sRed);
        if (tid == 0) partials[bofs + e] = s;
    }
#pragma unroll
    for (int e = 0; e < N_EXP; ++e) {
        const float s = block_reduce((i1 == e) ? 1.0f : 0.0f, sRed);
        if (tid == 0) partials[bofs + 8 + e] = s;
    }
    {
        const float s = block_reduce(z * z, sRed);
        if (tid == 0) partials[bofs + 16] = s;
    }
}

__global__ void finalize_kernel(const float* __restrict__ partials,
                                int nblocks, float* __restrict__ out_tail)
{
    if (threadIdx.x == 0 && blockIdx.x == 0) {
        float imp[N_EXP], ld[N_EXP], z2 = 0.0f;
        for (int e = 0; e < N_EXP; ++e) { imp[e] = 0.0f; ld[e] = 0.0f; }
        for (int b = 0; b < nblocks; ++b) {
            for (int e = 0; e < N_EXP; ++e) {
                imp[e] += partials[b * 32 + e];
                ld[e]  += partials[b * 32 + 8 + e];
            }
            z2 += partials[b * 32 + 16];
        }
        const float inv_t = 1.0f / (float)T_TOK;
        float aux = 0.0f;
        for (int e = 0; e < N_EXP; ++e) aux += (imp[e] * inv_t) * (ld[e] * inv_t);
        out_tail[0] = ZLOSS_COEF * z2 * inv_t;   // router_loss
        out_tail[1] = aux * (float)N_EXP;        // aux_loss
    }
}

// ---------------------------------------------------------------------------
extern "C" void kernel_launch(void* const* d_in, const int* in_sizes, int n_in,
                              void* d_out, int out_size, void* d_ws, size_t ws_size,
                              hipStream_t stream)
{
    (void)in_sizes; (void)n_in; (void)out_size; (void)ws_size;

    const float* x  = (const float*)d_in[0];
    const float* Wr = (const float*)d_in[1];
    const float* W1 = (const float*)d_in[2];
    const float* W3 = (const float*)d_in[3];
    const float* W2 = (const float*)d_in[4];
    float* out = (float*)d_out;

    // workspace layout (~33.9 MB):
    //   hbuf:    T * D_FF bf16              = 33,554,432 B
    //   gates:   T * N_EXP f32              =    262,144 B
    //   partials:32 blocks * 32 f32         =      4,096 B
    unsigned short* hbuf  = (unsigned short*)d_ws;
    float* gates    = (float*)((char*)d_ws + (size_t)T_TOK * D_FF * 2);
    float* partials = gates + (size_t)T_TOK * N_EXP;

    const int nblk = T_TOK / 256;   // 32 router blocks

    router_kernel<<<nblk, 256, 0, stream>>>(x, Wr, gates, partials);

    for (int e = 0; e < N_EXP; ++e) {
        const float* W1e = W1 + (size_t)e * D_FF * D_MODEL;
        const float* W3e = W3 + (size_t)e * D_FF * D_MODEL;
        const float* W2e = W2 + (size_t)e * D_MODEL * D_FF;

        dim3 g1(T_TOK / 128, D_FF / 64);     // 64 x 32
        ffn_h_kernel<<<g1, 256, 0, stream>>>(x, W1e, W3e, hbuf);

        dim3 g2(T_TOK / 128, D_MODEL / 64);  // 64 x 16
        ffn_y_kernel<<<g2, 256, 0, stream>>>(hbuf, W2e, gates, out, e, e == 0);
    }

    finalize_kernel<<<1, 32, 0, stream>>>(partials, nblk, out + (size_t)T_TOK * D_MODEL);
}
